// GNN_8830452760603
// MI455X (gfx1250) — matrix operand order repaired and checked by reading, not compiled, not run on hardware
//
#include <hip/hip_runtime.h>
#include <math.h>

// ---------------------------------------------------------------------------
// GNN (2x GraphConv) for MI455X / gfx1250.
// Densified normalized adjacency (4096x4096 bf16, L2-resident) turns the
// whole network into 4 bf16 WMMA GEMMs with f32 accumulation.
// GEMM: 256x128 workgroup tile, 8 waves @ 64x64, K-tile 32, double-buffered
// LDS fed by gfx1250 async DMA (global_load_async_to_lds_b128 / ASYNCcnt).
// ---------------------------------------------------------------------------

typedef __attribute__((ext_vector_type(16))) __bf16 v16bf;
typedef __attribute__((ext_vector_type(8)))  float  v8f;

#define NN   4096      // nodes
#define EE   131072    // edges
#define FIN  512       // input features
#define FH   8192      // hidden
#define NCLS 16        // classes

static __device__ __forceinline__ unsigned short f2bf(float f) {
  union { float f; unsigned u; } v; v.f = f;
  unsigned r = v.u + 0x7FFFu + ((v.u >> 16) & 1u);   // round-to-nearest-even
  return (unsigned short)(r >> 16);
}
static __device__ __forceinline__ float bf2f(unsigned short h) {
  union { unsigned u; float f; } v; v.u = ((unsigned)h) << 16;
  return v.f;
}

// --------------------------- prolog kernels --------------------------------

__global__ void k_count_deg(const int* __restrict__ src, const int* __restrict__ dst,
                            float* __restrict__ degO, float* __restrict__ degI, int E) {
  int i = blockIdx.x * blockDim.x + threadIdx.x;
  if (i < E) {
    atomicAdd(&degO[src[i]], 1.0f);
    atomicAdd(&degI[dst[i]], 1.0f);
  }
}

__global__ void k_invsqrt(float* __restrict__ a, float* __restrict__ b, int n) {
  int i = blockIdx.x * blockDim.x + threadIdx.x;
  if (i < n) {
    a[i] = rsqrtf(fmaxf(a[i], 1.0f));
    b[i] = rsqrtf(fmaxf(b[i], 1.0f));
  }
}

__global__ void k_adj_count(const int* __restrict__ src, const int* __restrict__ dst,
                            float* __restrict__ Acnt, int E, int n) {
  int i = blockIdx.x * blockDim.x + threadIdx.x;
  if (i < E) atomicAdd(&Acnt[(size_t)dst[i] * n + src[i]], 1.0f);
}

// Abf[d,s] = count(d,s) * invI[d] * invO[s]   (folds both norm scalings)
__global__ void k_adj_bf16(const float* __restrict__ Acnt,
                           const float* __restrict__ invI, const float* __restrict__ invO,
                           unsigned short* __restrict__ Abf, int n) {
  int i = blockIdx.x * blockDim.x + threadIdx.x;
  int total = n * n;
  if (i < total) {
    int r = i / n, c = i - r * n;
    Abf[i] = f2bf(Acnt[i] * invI[r] * invO[c]);
  }
}

__global__ void k_f32_to_bf16(const float* __restrict__ s, unsigned short* __restrict__ d, int n) {
  int i = blockIdx.x * blockDim.x + threadIdx.x;
  if (i < n) d[i] = f2bf(s[i]);
}

// s: [M][Nc] row-major f32  ->  d: [Nc][M] row-major bf16  (i.e. transpose)
__global__ void k_f32_to_bf16_T(const float* __restrict__ s, unsigned short* __restrict__ d,
                                int M, int Nc) {
  int i = blockIdx.x * blockDim.x + threadIdx.x;
  if (i < M * Nc) {
    int m = i / Nc, n = i - m * Nc;
    d[(size_t)n * M + m] = f2bf(s[i]);
  }
}

// ------------------------------ WMMA GEMM ----------------------------------
// C = A * B (+bias) (+relu), bf16 in / f32 accumulate / bf16 out.
//   A   : [M][K] row-major
//   Bcm : B column-major, stored as [N][K] row-major
//   C   : STORE_T ? [N][M] (C^T) : [M][N]
// Workgroup tile 256(M) x 128(N), K-tile 32; 8 waves in 4(M) x 2(N) grid,
// each wave owns 64x64 = 4x4 fragments -> 16 v_wmma per K-step, fed by
// 16 ds_load_b128 (1:1). Global->LDS staging is async DMA (ASYNCcnt).

#define TSTR 48   // padded LDS row stride (elements) to dodge bank conflicts

template<bool STORE_T, bool RELU, bool BIAS>
__global__ __launch_bounds__(256)
void k_gemm_bf16(const unsigned short* __restrict__ A,
                 const unsigned short* __restrict__ Bcm,
                 const float* __restrict__ bias,
                 unsigned short* __restrict__ C,
                 int M, int Nn, int K)
{
  __shared__ __align__(16) unsigned short As[2][256 * TSTR];
  __shared__ __align__(16) unsigned short Bs[2][128 * TSTR];

  const int tid  = threadIdx.x;
  const int lane = tid & 31;
  const int wv   = tid >> 5;
  const int wm   = (wv & 3) * 64;   // wave row offset in tile (0..192)
  const int wn   = (wv >> 2) * 64;  // wave col offset in tile (0 or 64)
  const int mBase = blockIdx.y * 256;
  const int nBase = blockIdx.x * 128;

  // ---- async global->LDS staging: A tile 256x32 (1024 chunks of 16B),
  //      B tile 128x32 (512 chunks); 256 threads -> 4 + 2 chunks each.
  auto stage = [&](int kt, int buf) {
    const int kcol = kt << 5;
#pragma unroll
    for (int q = 0; q < 4; ++q) {
      const int c    = tid + (q << 8);           // 0..1023
      const int row  = c >> 2;
      const int col8 = (c & 3) << 3;
      unsigned long long ga =
          (unsigned long long)(size_t)(A + (size_t)(mBase + row) * K + kcol + col8);
      unsigned lofs = (unsigned)(size_t)(const void*)&As[buf][row * TSTR + col8];
      asm volatile("global_load_async_to_lds_b128 %0, %1, off"
                   :: "v"(lofs), "v"(ga) : "memory");
    }
#pragma unroll
    for (int q = 0; q < 2; ++q) {
      const int c    = tid + (q << 8);           // 0..511
      const int row  = c >> 2;
      const int col8 = (c & 3) << 3;
      unsigned long long ga =
          (unsigned long long)(size_t)(Bcm + (size_t)(nBase + row) * K + kcol + col8);
      unsigned lofs = (unsigned)(size_t)(const void*)&Bs[buf][row * TSTR + col8];
      asm volatile("global_load_async_to_lds_b128 %0, %1, off"
                   :: "v"(lofs), "v"(ga) : "memory");
    }
  };

  const v8f vzero = {0.f, 0.f, 0.f, 0.f, 0.f, 0.f, 0.f, 0.f};
  v8f acc[4][4];
#pragma unroll
  for (int i = 0; i < 4; ++i)
#pragma unroll
    for (int j = 0; j < 4; ++j) acc[i][j] = vzero;

  union Frag { v16bf v; uint4 q[2]; };

  const int lr = lane & 15;
  const int kA = (lane < 16) ? 0 : 8;   // A: lanes16-31 hold K=8..15,24..31
  const int kB = (lane < 16) ? 0 : 8;   // B: lanes16-31 hold K=8..15,24..31

  stage(0, 0);
  asm volatile("s_wait_asynccnt 0x0" ::: "memory");
  __syncthreads();

  const int KT = K >> 5;
  for (int kt = 0; kt < KT; ++kt) {
    const int cur = kt & 1;
    if (kt + 1 < KT) stage(kt + 1, cur ^ 1);   // DMA next tile while computing

    Frag af[4], bg[4];
#pragma unroll
    for (int i = 0; i < 4; ++i) {
      const unsigned short* p = &As[cur][(wm + i * 16 + lr) * TSTR];
      af[i].q[0] = *(const uint4*)(p + kA);
      af[i].q[1] = *(const uint4*)(p + kA + 16);
    }
#pragma unroll
    for (int j = 0; j < 4; ++j) {
      const unsigned short* p = &Bs[cur][(wn + j * 16 + lr) * TSTR];
      bg[j].q[0] = *(const uint4*)(p + kB);
      bg[j].q[1] = *(const uint4*)(p + kB + 16);
    }

#pragma unroll
    for (int i = 0; i < 4; ++i)
#pragma unroll
      for (int j = 0; j < 4; ++j)
        acc[i][j] = __builtin_amdgcn_wmma_f32_16x16x32_bf16(
            false, af[i].v, false, bg[j].v, (short)0, acc[i][j], false, false);

    asm volatile("s_wait_asynccnt 0x0" ::: "memory");   // DMA done before barrier
    __syncthreads();
  }

  // epilogue: C/D layout => lane holds column n = 16*frag + lane%16,
  // rows m = 8*(lane/16) + v  (v = accumulator element index, contiguous)
#pragma unroll
  for (int i = 0; i < 4; ++i) {
#pragma unroll
    for (int j = 0; j < 4; ++j) {
      v8f c = acc[i][j];
      const int n  = nBase + wn + j * 16 + lr;
      const int m0 = mBase + wm + i * 16 + ((lane >> 4) << 3);
      if (BIAS) {
        float bv = bias[n];
#pragma unroll
        for (int v = 0; v < 8; ++v) c[v] += bv;
      }
      if (RELU) {
#pragma unroll
        for (int v = 0; v < 8; ++v) c[v] = fmaxf(c[v], 0.0f);
      }
      if (STORE_T) {
        union { unsigned short s[8]; uint4 q; } t;
#pragma unroll
        for (int v = 0; v < 8; ++v) t.s[v] = f2bf(c[v]);
        *(uint4*)&C[(size_t)n * M + m0] = t.q;      // 16B contiguous per lane
      } else {
#pragma unroll
        for (int v = 0; v < 8; ++v)
          C[(size_t)(m0 + v) * Nn + n] = f2bf(c[v]);
      }
    }
  }
}

// --------------------- final: h @ Wl + bl, softmax(16) ---------------------

__global__ __launch_bounds__(256)
void k_final(const unsigned short* __restrict__ h, const float* __restrict__ Wl,
             const float* __restrict__ bl, float* __restrict__ out, int n)
{
  __shared__ float s[NCLS];
  const int row = blockIdx.x;
  const int tid = threadIdx.x;
  if (tid < NCLS) s[tid] = 0.0f;
  __syncthreads();

  float acc[NCLS];
#pragma unroll
  for (int c = 0; c < NCLS; ++c) acc[c] = 0.0f;

  for (int i = tid; i < n; i += blockDim.x) {
    float x = bf2f(h[(size_t)row * n + i]);
#pragma unroll
    for (int c = 0; c < NCLS; ++c) acc[c] += x * Wl[i * NCLS + c];
  }
#pragma unroll
  for (int c = 0; c < NCLS; ++c) atomicAdd(&s[c], acc[c]);
  __syncthreads();

  if (tid == 0) {
    float v[NCLS], mx = -1e30f;
#pragma unroll
    for (int c = 0; c < NCLS; ++c) { v[c] = s[c] + bl[c]; mx = fmaxf(mx, v[c]); }
    float sum = 0.0f;
#pragma unroll
    for (int c = 0; c < NCLS; ++c) { v[c] = expf(v[c] - mx); sum += v[c]; }
    float inv = 1.0f / sum;
#pragma unroll
    for (int c = 0; c < NCLS; ++c) out[(size_t)row * NCLS + c] = v[c] * inv;
  }
}

// ------------------------------ launcher -----------------------------------

extern "C" void kernel_launch(void* const* d_in, const int* in_sizes, int n_in,
                              void* d_out, int out_size, void* d_ws, size_t ws_size,
                              hipStream_t stream)
{
  (void)in_sizes; (void)n_in; (void)out_size; (void)ws_size;

  const float* feat = (const float*)d_in[0];   // [4096][512]
  const float* W1   = (const float*)d_in[1];   // [512][8192]
  const float* b1   = (const float*)d_in[2];   // [8192]
  const float* W2   = (const float*)d_in[3];   // [8192][4096]
  const float* Wl   = (const float*)d_in[4];   // [4096][16]
  const float* bl   = (const float*)d_in[5];   // [16]
  const int*   src  = (const int*)d_in[6];     // [E]
  const int*   dst  = (const int*)d_in[7];     // [E]
  float* out = (float*)d_out;                  // [4096][16]

  char* ws = (char*)d_ws;
  size_t off = 0;
  auto alloc = [&](size_t bytes) -> void* {
    void* p = ws + off;
    off = (off + bytes + 255) & ~(size_t)255;
    return p;
  };

  float*          invO  = (float*)alloc((size_t)NN * 4);            // deg_out -> rsqrt
  float*          invI  = (float*)alloc((size_t)NN * 4);            // deg_in  -> rsqrt
  float*          Acnt  = (float*)alloc((size_t)NN * NN * 4);
  unsigned short* Abf   = (unsigned short*)alloc((size_t)NN * NN * 2);
  unsigned short* featb = (unsigned short*)alloc((size_t)NN * FIN * 2);
  unsigned short* W1t   = (unsigned short*)alloc((size_t)FH * FIN * 2);   // [8192][512]
  unsigned short* W2t   = (unsigned short*)alloc((size_t)NN * FH * 2);    // [4096][8192]
  unsigned short* x1t   = (unsigned short*)alloc((size_t)FH * NN * 2);    // x1^T
  unsigned short* agg1  = (unsigned short*)alloc((size_t)NN * FH * 2);    // row-major
  unsigned short* x2t   = (unsigned short*)alloc((size_t)NN * NN * 2);    // x2^T
  unsigned short* hbuf  = (unsigned short*)alloc((size_t)NN * NN * 2);    // relu'd, row-major

  hipMemsetAsync(invO, 0, (size_t)NN * 4, stream);
  hipMemsetAsync(invI, 0, (size_t)NN * 4, stream);
  hipMemsetAsync(Acnt, 0, (size_t)NN * NN * 4, stream);

  k_count_deg<<<(EE + 255) / 256, 256, 0, stream>>>(src, dst, invO, invI, EE);
  k_invsqrt<<<(NN + 255) / 256, 256, 0, stream>>>(invO, invI, NN);
  k_adj_count<<<(EE + 255) / 256, 256, 0, stream>>>(src, dst, Acnt, EE, NN);
  k_adj_bf16<<<((NN * NN) + 255) / 256, 256, 0, stream>>>(Acnt, invI, invO, Abf, NN);

  k_f32_to_bf16<<<((NN * FIN) + 255) / 256, 256, 0, stream>>>(feat, featb, NN * FIN);
  k_f32_to_bf16_T<<<((FIN * FH) + 255) / 256, 256, 0, stream>>>(W1, W1t, FIN, FH);
  k_f32_to_bf16_T<<<((FH * NN) + 255) / 256, 256, 0, stream>>>(W2, W2t, FH, NN);

  // gemm1: x1^T = (feat @ W1 + b1)^T       M=4096 N=8192 K=512
  k_gemm_bf16<true,  false, true ><<<dim3(FH / 128, NN / 256), 256, 0, stream>>>(
      featb, W1t, b1, x1t, NN, FH, FIN);
  // gemm2: agg1 = Abf @ x1                 M=4096 N=8192 K=4096 (B cm = x1t)
  k_gemm_bf16<false, false, false><<<dim3(FH / 128, NN / 256), 256, 0, stream>>>(
      Abf, x1t, nullptr, agg1, NN, FH, NN);
  // gemm3: x2^T = (agg1 @ W2)^T            M=4096 N=4096 K=8192 (B cm = W2t)
  k_gemm_bf16<true,  false, false><<<dim3(NN / 128, NN / 256), 256, 0, stream>>>(
      agg1, W2t, nullptr, x2t, NN, NN, FH);
  // gemm4: h = relu(Abf @ x2)              M=4096 N=4096 K=4096 (B cm = x2t)
  k_gemm_bf16<false, true,  false><<<dim3(NN / 128, NN / 256), 256, 0, stream>>>(
      Abf, x2t, nullptr, hbuf, NN, NN, NN);

  // final: softmax(h @ Wl + bl)
  k_final<<<NN, 256, 0, stream>>>(hbuf, Wl, bl, out, NN);
}
